// InputAttention_23922967839340
// MI455X (gfx1250) — matrix operand order repaired and checked by reading, not compiled, not run on hardware
//
#include <hip/hip_runtime.h>
#include <hip/hip_bf16.h>
#include <math.h>

// ---------------------------------------------------------------------------
// Problem constants (from reference): T=512, N=512, M=256, B=256
// ---------------------------------------------------------------------------
#define TT 512
#define NN 512
#define MM 256
#define BB 256

#define NCHUNK 128   // n values per workgroup (main kernel)
#define SCHUNK 128   // s values staged per LDS pass
#define PITCH  520   // bf16 elements per LDS row (512 + 8 pad -> bank stride 4)

typedef __bf16 v16bf __attribute__((ext_vector_type(16)));
typedef __bf16 v8bf  __attribute__((ext_vector_type(8)));
typedef float  v8f   __attribute__((ext_vector_type(8)));
typedef int    v4i   __attribute__((ext_vector_type(4)));

#if __has_builtin(__builtin_amdgcn_global_load_async_to_lds_b128) && \
    __has_builtin(__builtin_amdgcn_s_wait_asynccnt)
#define USE_ASYNC_LDS 1
#else
#define USE_ASYNC_LDS 0
#endif

static __device__ __forceinline__ unsigned short f32_to_bf16(float f) {
    unsigned int u = __builtin_bit_cast(unsigned int, f);
    u += 0x7FFFu + ((u >> 16) & 1u);          // round-to-nearest-even
    return (unsigned short)(u >> 16);
}

// Branch-free tanh: prefer hardware V_TANH_F32 (gfx1250 TRANS op).
#if __has_builtin(__builtin_amdgcn_tanhf)
static __device__ __forceinline__ float fast_tanh(float x) {
    return __builtin_amdgcn_tanhf(x);
}
#else
static __device__ __forceinline__ float fast_tanh(float x) {
    // tanh(x) = sign(x) * (1 - e^(-2|x|)) / (1 + e^(-2|x|)); e^(-2|x|) via v_exp_f32
    float ax = __builtin_fabsf(x);
    float e  = __builtin_amdgcn_exp2f(ax * -2.8853900817779268f);
    float r  = (1.0f - e) * __builtin_amdgcn_rcpf(1.0f + e);
    return __builtin_copysignf(r, x);
}
#endif

// ---------------------------------------------------------------------------
// Kernel U: convert U_e (512x512 f32, row s major) -> bf16 scratch.
// Row s of this buffer is column s of the WMMA B matrix (B[t,s] = U_e[s,t]).
// ---------------------------------------------------------------------------
__global__ __launch_bounds__(256) void uconv_kernel(
    const float* __restrict__ U_e, unsigned short* __restrict__ Ubf)
{
    for (int idx = blockIdx.x * blockDim.x + threadIdx.x;
         idx < TT * TT; idx += gridDim.x * blockDim.x)
        Ubf[idx] = f32_to_bf16(U_e[idx]);
}

// ---------------------------------------------------------------------------
// Kernel A: wq[b,s] = sum_k concat(h,s)[b,k] * W_e[s,k]   (full fp32)
// grid (8 s-chunks of 64, 16 b-tiles of 16), 256 threads = 16b x 16s
// ---------------------------------------------------------------------------
__global__ __launch_bounds__(256) void wq_kernel(
    const float* __restrict__ h_t, const float* __restrict__ s_t,
    const float* __restrict__ W_e, float* __restrict__ wq)
{
    __shared__ float qs[16 * 512];      // queries: broadcast reads, no pad needed
    __shared__ float wrow[16 * 520];    // W rows: pad 520 -> 8-bank stride

    const int b0  = blockIdx.y * 16;
    const int s0b = blockIdx.x * 64;
    const int tid = threadIdx.x;

    for (int idx = tid; idx < 16 * 512; idx += 256) {
        int bi = idx >> 9, k = idx & 511;
        float v = (k < MM) ? h_t[(b0 + bi) * MM + k]
                           : s_t[(b0 + bi) * MM + (k - MM)];
        qs[bi * 512 + k] = v;
    }

    const int bi = tid >> 4;
    const int si = tid & 15;

    for (int s0 = s0b; s0 < s0b + 64; s0 += 16) {
        __syncthreads();
        for (int idx = tid; idx < 16 * 512; idx += 256) {
            int r = idx >> 9, k = idx & 511;
            wrow[r * 520 + k] = W_e[(size_t)(s0 + r) * 512 + k];
        }
        __syncthreads();
        float acc = 0.0f;
        #pragma unroll 8
        for (int k = 0; k < 512; ++k)
            acc = fmaf(qs[bi * 512 + k], wrow[si * 520 + k], acc);
        wq[(b0 + bi) * TT + (s0 + si)] = acc;
    }
}

// ---------------------------------------------------------------------------
// Main kernel: fused  ud = data[b]^T @ U_e^T  ->  v . tanh(wq + ud)  -> scores
// grid (4 n-chunks, 256 b), 256 threads (8 wave32).
// LDS: A tile [128 n][512 t] bf16 + B tile [128 s][512 t] bf16 (+pad) ~ 260 KB
// Each wave: 4 n-tiles x 2 s-tiles of 16x16, K-loop of v_wmma_f32_16x16x32_bf16.
// ---------------------------------------------------------------------------
__global__ __launch_bounds__(256) void attn_scores_kernel(
    const float* __restrict__ data, const unsigned short* __restrict__ Ubf,
    const float* __restrict__ wq, const float* __restrict__ v_e,
    float* __restrict__ scores)
{
    __shared__ __align__(16) unsigned short sA[NCHUNK * PITCH];
    __shared__ __align__(16) unsigned short sB[SCHUNK * PITCH];
    __shared__ float sScore[NCHUNK];

    const int b    = blockIdx.y;
    const int n0   = blockIdx.x * NCHUNK;
    const int tid  = threadIdx.x;
    const int lane = tid & 31;
    const int wave = tid >> 5;

    if (tid < NCHUNK) sScore[tid] = 0.0f;

    // ---- Stage A: sA[n][t] = bf16(data[b, t, n0+n])  (coalesced float4 reads)
    {
        const float* src = data + (size_t)b * TT * NN + n0;
        for (int idx = tid; idx < TT * (NCHUNK / 4); idx += 256) {
            int t  = idx >> 5;          // 0..511
            int c4 = idx & 31;          // float4 column within the 128-wide row
            const float4 f = *(const float4*)(src + (size_t)t * NN + c4 * 4);
            int n = c4 * 4;
            sA[(n + 0) * PITCH + t] = f32_to_bf16(f.x);
            sA[(n + 1) * PITCH + t] = f32_to_bf16(f.y);
            sA[(n + 2) * PITCH + t] = f32_to_bf16(f.z);
            sA[(n + 3) * PITCH + t] = f32_to_bf16(f.w);
        }
    }

    const int m16  = lane & 15;
    const int half = lane >> 4;
    const int nt0  = (wave & 1) * 4;    // 4 n-tiles per wave
    const int st0  = (wave >> 1) * 2;   // 2 s-tiles per wave

    const v8f vzero = {0.f, 0.f, 0.f, 0.f, 0.f, 0.f, 0.f, 0.f};

    for (int sc = 0; sc < TT / SCHUNK; ++sc) {
        __syncthreads();   // previous pass fully consumed sB (and sA staged)

        // ---- Stage B chunk: sB[sl][t] = Ubf[(sc*128+sl)*512 + t]
#if USE_ASYNC_LDS
        {
            // CDNA5 async copy: global -> LDS without VGPR round trip (ASYNCcnt)
            const unsigned short* src = Ubf + (size_t)(sc * SCHUNK) * TT;
            for (int idx = tid; idx < SCHUNK * (TT / 8); idx += 256) {
                int sl = idx >> 6;
                int c8 = idx & 63;
                __builtin_amdgcn_global_load_async_to_lds_b128(
                    (v4i*)(src + idx * 8),
                    (v4i*)&sB[sl * PITCH + c8 * 8],
                    0, 0);
            }
            __builtin_amdgcn_s_wait_asynccnt(0);
        }
#else
        {
            const uint4* src = (const uint4*)(Ubf + (size_t)(sc * SCHUNK) * TT);
            for (int idx = tid; idx < SCHUNK * (TT / 8); idx += 256) {
                int sl = idx >> 6;
                int c8 = idx & 63;
                *(uint4*)&sB[sl * PITCH + c8 * 8] = src[idx];
            }
        }
#endif
        __syncthreads();

        // ---- WMMA: acc[i][j] (16n x 16s) += A(16x32) x B(32x16) over K=512
        v8f acc[4][2];
        #pragma unroll
        for (int i = 0; i < 4; ++i)
            #pragma unroll
            for (int j = 0; j < 2; ++j) acc[i][j] = vzero;

        const unsigned short* aB[4];
        #pragma unroll
        for (int i = 0; i < 4; ++i)
            aB[i] = &sA[((nt0 + i) * 16 + m16) * PITCH];
        const unsigned short* bB[2];
        #pragma unroll
        for (int j = 0; j < 2; ++j)
            bB[j] = &sB[((st0 + j) * 16 + m16) * PITCH];

        for (int k0 = 0; k0 < TT; k0 += 32) {
            v16bf aF[4], bF[2];
            #pragma unroll
            for (int i = 0; i < 4; ++i) {
                // A layout (ISA 7.12.2): lane row m=lane%16; elems 0-7 -> K=8*half+e,
                // elems 8-15 -> K=16+8*half+(e-8)
                v8bf lo = *(const v8bf*)(aB[i] + k0 + 8 * half);
                v8bf hi = *(const v8bf*)(aB[i] + k0 + 16 + 8 * half);
                aF[i] = __builtin_shufflevector(lo, hi,
                        0,1,2,3,4,5,6,7,8,9,10,11,12,13,14,15);
            }
            #pragma unroll
            for (int j = 0; j < 2; ++j) {
                // B layout: lane col n=lane%16; elems 0-7 -> K=8*half+e, elems 8-15 -> K=16+8*half+(e-8)
                v8bf lo = *(const v8bf*)(bB[j] + k0 + 8 * half);
                v8bf hi = *(const v8bf*)(bB[j] + k0 + 16 + 8 * half);
                bF[j] = __builtin_shufflevector(lo, hi,
                        0,1,2,3,4,5,6,7,8,9,10,11,12,13,14,15);
            }
            #pragma unroll
            for (int i = 0; i < 4; ++i)
                #pragma unroll
                for (int j = 0; j < 2; ++j)
                    acc[i][j] = __builtin_amdgcn_wmma_f32_16x16x32_bf16(
                        false, aF[i], false, bF[j],
                        (short)0, acc[i][j], false, false);
        }

        // ---- Fused epilogue: score[n] += sum_s v[s] * tanh(wq[b,s] + ud[n,s])
        // C/D layout: lane holds column s = lane%16; VGPR r holds row r + 8*half.
        // Combine both s-tiles per wave in registers -> one ds_add_f32 per row.
        {
            int s0g = sc * SCHUNK + st0 * 16 + m16;
            float wqs0 = wq[b * TT + s0g];
            float vs0  = v_e[s0g];
            float wqs1 = wq[b * TT + s0g + 16];
            float vs1  = v_e[s0g + 16];
            #pragma unroll
            for (int i = 0; i < 4; ++i) {
                int nb = (nt0 + i) * 16 + 8 * half;
                #pragma unroll
                for (int r = 0; r < 8; ++r) {
                    float val = vs0 * fast_tanh(wqs0 + acc[i][0][r])
                              + vs1 * fast_tanh(wqs1 + acc[i][1][r]);
                    atomicAdd(&sScore[nb + r], val);   // ds_add_f32
                }
            }
        }
    }

    __syncthreads();
    if (tid < NCHUNK)
        scores[(size_t)b * NN + n0 + tid] = sScore[tid];
}

// ---------------------------------------------------------------------------
// Kernel C: softmax over n (512) per batch row; out shape (B,1,N) flat.
// ---------------------------------------------------------------------------
__global__ __launch_bounds__(256) void softmax_kernel(
    const float* __restrict__ scores, float* __restrict__ out)
{
    __shared__ float red[256];
    const int b   = blockIdx.x;
    const int tid = threadIdx.x;

    float x0 = scores[b * NN + tid];
    float x1 = scores[b * NN + 256 + tid];

    red[tid] = fmaxf(x0, x1);
    __syncthreads();
    for (int s = 128; s > 0; s >>= 1) {
        if (tid < s) red[tid] = fmaxf(red[tid], red[tid + s]);
        __syncthreads();
    }
    float m = red[0];
    __syncthreads();

    float e0 = expf(x0 - m), e1 = expf(x1 - m);
    red[tid] = e0 + e1;
    __syncthreads();
    for (int s = 128; s > 0; s >>= 1) {
        if (tid < s) red[tid] += red[tid + s];
        __syncthreads();
    }
    float inv = 1.0f / red[0];
    out[b * NN + tid]       = e0 * inv;
    out[b * NN + 256 + tid] = e1 * inv;
}

// ---------------------------------------------------------------------------
// Launch: inputs in setup order: h_t, s_t, data, W_e, U_e, v_e
// ws: [wq f32 131072][scores f32 131072][Ubf bf16 262144]  = 1.5 MB
// ---------------------------------------------------------------------------
extern "C" void kernel_launch(void* const* d_in, const int* in_sizes, int n_in,
                              void* d_out, int out_size, void* d_ws, size_t ws_size,
                              hipStream_t stream) {
    const float* h_t  = (const float*)d_in[0];
    const float* s_t  = (const float*)d_in[1];
    const float* data = (const float*)d_in[2];
    const float* W_e  = (const float*)d_in[3];
    const float* U_e  = (const float*)d_in[4];
    const float* v_e  = (const float*)d_in[5];
    float* out = (float*)d_out;

    float* wq              = (float*)d_ws;              // 131072 f32
    float* scores          = wq + BB * TT;              // 131072 f32
    unsigned short* Ubf    = (unsigned short*)(scores + BB * NN); // 262144 bf16

    uconv_kernel<<<256, 256, 0, stream>>>(U_e, Ubf);
    wq_kernel<<<dim3(8, 16), 256, 0, stream>>>(h_t, s_t, W_e, wq);
    attn_scores_kernel<<<dim3(NN / NCHUNK, BB), 256, 0, stream>>>(
        data, Ubf, wq, v_e, scores);
    softmax_kernel<<<BB, 256, 0, stream>>>(scores, out);
}
